// MultiHeadAttention_4243427688919
// MI455X (gfx1250) — matrix operand order repaired and checked by reading, not compiled, not run on hardware
//
#include <hip/hip_runtime.h>
#include <hip/hip_bf16.h>

// ---------------------------------------------------------------------------
// MHA forward for MI455X (gfx1250, wave32, WMMA bf16 16x16x32, f32 accum)
// B=4, S=1024, D=1024, H=16, DK=DH=64
// ---------------------------------------------------------------------------

#define BB  4
#define SS  1024
#define DDIM 1024
#define HH  16
#define DKK 64
#define EE  1024            // H*DK
#define MM  (BB*SS)         // 4096 rows in the big GEMMs

typedef __attribute__((ext_vector_type(16))) __bf16 v16bf;
typedef __attribute__((ext_vector_type(8)))  float  v8f;

union BF16x16 {
    v16bf v;
    unsigned short h[16];
    uint4 q[2];
};

static __device__ inline unsigned short f2bf(float f) {
    union { float f; unsigned u; } x; x.f = f;
    unsigned u = x.u;
    u += 0x7fffu + ((u >> 16) & 1u);   // round to nearest even
    return (unsigned short)(u >> 16);
}

// B fragment (32x16, bf16): lane n<16 holds (K=0..15, N=n), lane n+16 holds K=16..31.
static __device__ inline v16bf load_b_frag(const unsigned short* p) {
    BF16x16 r;
    r.q[0] = *(const uint4*)(p);
    r.q[1] = *(const uint4*)(p + 8);
    return r.v;
}

// A fragment (16x32, bf16): lane m<16 holds K=0..15 of row m;
// lane m+16 holds K=16..31 of row m.
static __device__ inline v16bf load_a_frag_bf16(const unsigned short* row, int hi) {
    BF16x16 r;
    r.q[0] = *(const uint4*)(row + hi * 16);
    r.q[1] = *(const uint4*)(row + hi * 16 + 8);
    return r.v;
}

static __device__ inline v8f wmma_bf16(v16bf a, v16bf b, v8f c) {
    return __builtin_amdgcn_wmma_f32_16x16x32_bf16(false, a, false, b, (short)0, c,
                                                   false, false);
}

// ---------------------------------------------------------------------------
// Kernel 1: LDS-tiled transpose + convert: Wt[n][k] = bf16(W[k][n])
// ---------------------------------------------------------------------------
__global__ __launch_bounds__(256) void wtrans_kernel(const float* __restrict__ W,
                                                     unsigned short* __restrict__ Wt) {
    __shared__ unsigned short tile[32][33];
    const int tx = threadIdx.x & 31;
    const int ty = threadIdx.x >> 5;
    const int kt = blockIdx.x * 32;
    const int nt = blockIdx.y * 32;
#pragma unroll
    for (int i = 0; i < 32; i += 8)
        tile[ty + i][tx] = f2bf(W[(unsigned)((kt + ty + i) * DDIM + nt + tx)]);
    __syncthreads();
#pragma unroll
    for (int i = 0; i < 32; i += 8)
        Wt[(unsigned)((nt + ty + i) * DDIM + kt + tx)] = tile[tx][ty + i];
}

// ---------------------------------------------------------------------------
// Kernel 2: vectorized f32 -> bf16 conversion (8 elements / thread)
// ---------------------------------------------------------------------------
__global__ __launch_bounds__(256) void cvt_bf16_kernel(const float* __restrict__ in,
                                                       unsigned short* __restrict__ out) {
    const unsigned i = ((unsigned)blockIdx.x * 256 + threadIdx.x) * 8;
    const float4 a = *(const float4*)(in + i);
    const float4 b = *(const float4*)(in + i + 4);
    uint4 o;
    o.x = (unsigned)f2bf(a.x) | ((unsigned)f2bf(a.y) << 16);
    o.y = (unsigned)f2bf(a.z) | ((unsigned)f2bf(a.w) << 16);
    o.z = (unsigned)f2bf(b.x) | ((unsigned)f2bf(b.y) << 16);
    o.w = (unsigned)f2bf(b.z) | ((unsigned)f2bf(b.w) << 16);
    *(uint4*)(out + i) = o;
}

// ---------------------------------------------------------------------------
// Kernel 3: projection GEMM  Y = X(bf16) * Wt(bf16)^T + bias
//   mode 0: store bf16 Y[m][n]                              (Q, K projections)
//   mode 1: store bf16 transposed per head: Vt[b][h][dh][s] (V projection)
// 8 waves in a 4x2 grid; wave tile 32(M) x 64(N); K-step 32, ping-pong
// double buffered (K unrolled by 2). Per K-step: 4 A + 8 B loads -> 8 WMMAs.
// Peak live ~160 VGPRs: no extended-VGPR traffic.
// ---------------------------------------------------------------------------
__global__ __launch_bounds__(256) void proj_gemm_kernel(
    const unsigned short* __restrict__ X, const unsigned short* __restrict__ Wt,
    const float* __restrict__ bias, unsigned short* __restrict__ Out, int mode) {
    const int tid = threadIdx.x;
    const int wave = tid >> 5;
    const int lane = tid & 31;
    const int l = lane & 15, hi = lane >> 4;
    const int wave_m = wave >> 1;                  // 0..3
    const int wave_n = wave & 1;                   // 0..1
    const int m0 = blockIdx.x * 128 + wave_m * 32;
    const int n0 = blockIdx.y * 128 + wave_n * 64;

    v8f acc[2][4];
#pragma unroll
    for (int mi = 0; mi < 2; mi++)
#pragma unroll
        for (int ni = 0; ni < 4; ni++) acc[mi][ni] = (v8f)0.0f;

    unsigned aoff[2], boff[4];
#pragma unroll
    for (int mi = 0; mi < 2; mi++) aoff[mi] = (unsigned)(m0 + mi * 16 + l) * DDIM;
#pragma unroll
    for (int ni = 0; ni < 4; ni++)
        boff[ni] = (unsigned)(n0 + ni * 16 + l) * DDIM + hi * 16;

    // prologue: buffer A <- k = 0
    v16bf aA[2], bA[4];
#pragma unroll
    for (int mi = 0; mi < 2; mi++) aA[mi] = load_a_frag_bf16(X + aoff[mi], hi);
#pragma unroll
    for (int ni = 0; ni < 4; ni++) bA[ni] = load_b_frag(Wt + boff[ni]);

    for (int k0 = 0; k0 < DDIM - 64; k0 += 64) {
        v16bf aB[2], bB[4];
#pragma unroll
        for (int mi = 0; mi < 2; mi++)
            aB[mi] = load_a_frag_bf16(X + aoff[mi] + k0 + 32, hi);
#pragma unroll
        for (int ni = 0; ni < 4; ni++)
            bB[ni] = load_b_frag(Wt + boff[ni] + k0 + 32);
        __builtin_prefetch(X + aoff[0] + k0 + 512, 0, 3);   // global_prefetch_b8
        __builtin_prefetch(X + aoff[1] + k0 + 512, 0, 3);
#pragma unroll
        for (int mi = 0; mi < 2; mi++)
#pragma unroll
            for (int ni = 0; ni < 4; ni++)
                acc[mi][ni] = wmma_bf16(aA[mi], bA[ni], acc[mi][ni]);

#pragma unroll
        for (int mi = 0; mi < 2; mi++)
            aA[mi] = load_a_frag_bf16(X + aoff[mi] + k0 + 64, hi);
#pragma unroll
        for (int ni = 0; ni < 4; ni++)
            bA[ni] = load_b_frag(Wt + boff[ni] + k0 + 64);
#pragma unroll
        for (int mi = 0; mi < 2; mi++)
#pragma unroll
            for (int ni = 0; ni < 4; ni++)
                acc[mi][ni] = wmma_bf16(aB[mi], bB[ni], acc[mi][ni]);
    }
    {   // tail: last two K-steps
        v16bf aB[2], bB[4];
#pragma unroll
        for (int mi = 0; mi < 2; mi++)
            aB[mi] = load_a_frag_bf16(X + aoff[mi] + DDIM - 32, hi);
#pragma unroll
        for (int ni = 0; ni < 4; ni++)
            bB[ni] = load_b_frag(Wt + boff[ni] + DDIM - 32);
#pragma unroll
        for (int mi = 0; mi < 2; mi++)
#pragma unroll
            for (int ni = 0; ni < 4; ni++)
                acc[mi][ni] = wmma_bf16(aA[mi], bA[ni], acc[mi][ni]);
#pragma unroll
        for (int mi = 0; mi < 2; mi++)
#pragma unroll
            for (int ni = 0; ni < 4; ni++)
                acc[mi][ni] = wmma_bf16(aB[mi], bB[ni], acc[mi][ni]);
    }

#pragma unroll
    for (int ni = 0; ni < 4; ni++) {
        const int n = n0 + ni * 16 + l;
        const float bv = bias[n];
#pragma unroll
        for (int mi = 0; mi < 2; mi++) {
#pragma unroll
            for (int j = 0; j < 8; j++) {
                const int m = m0 + mi * 16 + j + hi * 8;
                unsigned short o = f2bf(acc[mi][ni][j] + bv);
                if (mode == 0) {
                    Out[(unsigned)(m * EE + n)] = o;
                } else {
                    const int bb = m >> 10, s = m & 1023;
                    const int h = n >> 6, dh = n & 63;
                    Out[(unsigned)((((bb * HH + h) * DKK + dh) * SS) + s)] = o;
                }
            }
        }
    }
}

// ---------------------------------------------------------------------------
// Kernel 4: flash attention. One wave per (b, h, 16-query tile).
// Online softmax over 64-key blocks; causal blocks skipped.
// ---------------------------------------------------------------------------
__global__ __launch_bounds__(256) void attn_kernel(
    const unsigned short* __restrict__ Qp, const unsigned short* __restrict__ Kp,
    const unsigned short* __restrict__ Vt, const int* __restrict__ v_mask,
    unsigned short* __restrict__ Oa) {
    __shared__ unsigned short plds[8][16 * 72];   // per-wave P staging (padded rows)

    const int tid = threadIdx.x;
    const int wave = tid >> 5, lane = tid & 31;
    const int l = lane & 15, hi = lane >> 4;

    const int ntq = SS / 16;                      // 64 query tiles per (b,h)
    const int wid = blockIdx.x * 8 + wave;        // 0 .. 4095
    const int b = wid / (HH * ntq);
    const int rem = wid % (HH * ntq);
    const int h = rem / ntq;
    const int q0 = (rem % ntq) * 16;

    // Q fragments (16 x 64 tile, two K=32 fragments), loaded once.
    const unsigned qoff = (unsigned)(b * SS + q0 + l) * EE + h * DKK;
    const v16bf aq0 = load_a_frag_bf16(Qp + qoff, hi);
    const v16bf aq1 = load_a_frag_bf16(Qp + qoff + 32, hi);

    float m_i[8], l_i[8];
#pragma unroll
    for (int j = 0; j < 8; j++) { m_i[j] = -1e30f; l_i[j] = 0.0f; }
    v8f acc[4];
#pragma unroll
    for (int t = 0; t < 4; t++) acc[t] = (v8f)0.0f;

    const float scale = 0.125f;                   // 1/sqrt(64)
    const int nblk = q0 / 64 + 1;                 // causal: keys 0 .. q0+15 only
    unsigned short* myl = &plds[wave][0];

    const unsigned koff = (unsigned)(b * SS + l) * EE + h * DKK + hi * 16;
    const unsigned voff = (unsigned)((b * HH + h) * DKK + l) * SS + hi * 16;

    for (int blk = 0; blk < nblk; blk++) {
        const int key0 = blk * 64;
        if (blk + 1 < nblk) {                      // prefetch next key block
            __builtin_prefetch(Kp + koff + (unsigned)(key0 + 64) * EE, 0, 3);
            __builtin_prefetch(Vt + voff + key0 + 64, 0, 3);
        }

        // ----- scores S = Q * K^T: load all 8 K fragments, then 8 WMMAs -----
        v16bf kb[8];
#pragma unroll
        for (int t = 0; t < 4; t++) {
            const unsigned o = koff + (unsigned)(key0 + t * 16) * EE;
            kb[2 * t]     = load_b_frag(Kp + o);
            kb[2 * t + 1] = load_b_frag(Kp + o + 32);
        }
        v8f s[4];
#pragma unroll
        for (int t = 0; t < 4; t++) {
            v8f z = (v8f)0.0f;
            z = wmma_bf16(aq0, kb[2 * t], z);
            z = wmma_bf16(aq1, kb[2 * t + 1], z);
            s[t] = z;
        }

        // additive key (validity) mask, per-lane key index
        float vadd[4];
#pragma unroll
        for (int t = 0; t < 4; t++) {
            const int n = key0 + t * 16 + l;
            vadd[t] = v_mask[b * SS + n] ? 0.0f : -1e12f;
        }

        // ----- online softmax per row (rows j and j+8 per lane half) -----
#pragma unroll
        for (int j = 0; j < 8; j++) {
            const int m = q0 + j + hi * 8;
            float p[4];
            float rmax = m_i[j];
#pragma unroll
            for (int t = 0; t < 4; t++) {
                const int n = key0 + t * 16 + l;
                float v = s[t][j] * scale + vadd[t];
                if (n > m) v -= 1e12f;            // causal mask
                p[t] = v;
                rmax = fmaxf(rmax, v);
            }
#pragma unroll
            for (int off = 8; off >= 1; off >>= 1)
                rmax = fmaxf(rmax, __shfl_xor(rmax, off, 16));

            const float f = __expf(m_i[j] - rmax);
            float rs = 0.0f;
#pragma unroll
            for (int t = 0; t < 4; t++) {
                const float e = __expf(p[t] - rmax);
                p[t] = e;
                rs += e;
            }
#pragma unroll
            for (int off = 8; off >= 1; off >>= 1)
                rs += __shfl_xor(rs, off, 16);

            l_i[j] = l_i[j] * f + rs;
            m_i[j] = rmax;
#pragma unroll
            for (int t = 0; t < 4; t++) acc[t][j] *= f;

            // stage P row into LDS (bf16) for A-fragment reload
            const int row = j + hi * 8;
#pragma unroll
            for (int t = 0; t < 4; t++)
                myl[row * 72 + t * 16 + l] = f2bf(p[t]);
        }

        // ----- O += P * V: load P/A and all 8 V fragments, then 8 WMMAs -----
        const v16bf ap0 = load_a_frag_bf16(myl + l * 72, hi);
        const v16bf ap1 = load_a_frag_bf16(myl + l * 72 + 32, hi);
        v16bf vb[8];
#pragma unroll
        for (int t = 0; t < 4; t++) {
            const unsigned o = voff + (unsigned)(t * 16) * SS + key0;
            vb[2 * t]     = load_b_frag(Vt + o);
            vb[2 * t + 1] = load_b_frag(Vt + o + 32);
        }
#pragma unroll
        for (int t = 0; t < 4; t++) {
            acc[t] = wmma_bf16(ap0, vb[2 * t], acc[t]);
            acc[t] = wmma_bf16(ap1, vb[2 * t + 1], acc[t]);
        }
    }

    // ----- normalize and store O tile (bf16) -----
#pragma unroll
    for (int t = 0; t < 4; t++) {
#pragma unroll
        for (int j = 0; j < 8; j++) {
            const int m = q0 + j + hi * 8;
            const float o = acc[t][j] / l_i[j];
            Oa[(unsigned)((b * SS + m) * EE + h * DKK + t * 16 + l)] = f2bf(o);
        }
    }
}

// ---------------------------------------------------------------------------
// Kernel 5: output GEMM  Y = Oa(bf16) * Wo^T(bf16) + bo, scaled by q_mask, f32 out
// Same 32x64 wave-tile, ping-pong pipelined.
// ---------------------------------------------------------------------------
__global__ __launch_bounds__(256) void out_gemm_kernel(
    const unsigned short* __restrict__ A, const unsigned short* __restrict__ Wt,
    const float* __restrict__ bias, const int* __restrict__ q_mask,
    float* __restrict__ Out) {
    const int tid = threadIdx.x;
    const int wave = tid >> 5;
    const int lane = tid & 31;
    const int l = lane & 15, hi = lane >> 4;
    const int wave_m = wave >> 1;
    const int wave_n = wave & 1;
    const int m0 = blockIdx.x * 128 + wave_m * 32;
    const int n0 = blockIdx.y * 128 + wave_n * 64;

    v8f acc[2][4];
#pragma unroll
    for (int mi = 0; mi < 2; mi++)
#pragma unroll
        for (int ni = 0; ni < 4; ni++) acc[mi][ni] = (v8f)0.0f;

    unsigned aoff[2], boff[4];
#pragma unroll
    for (int mi = 0; mi < 2; mi++) aoff[mi] = (unsigned)(m0 + mi * 16 + l) * EE;
#pragma unroll
    for (int ni = 0; ni < 4; ni++)
        boff[ni] = (unsigned)(n0 + ni * 16 + l) * EE + hi * 16;

    v16bf aA[2], bA[4];
#pragma unroll
    for (int mi = 0; mi < 2; mi++) aA[mi] = load_a_frag_bf16(A + aoff[mi], hi);
#pragma unroll
    for (int ni = 0; ni < 4; ni++) bA[ni] = load_b_frag(Wt + boff[ni]);

    for (int k0 = 0; k0 < EE - 64; k0 += 64) {
        v16bf aB[2], bB[4];
#pragma unroll
        for (int mi = 0; mi < 2; mi++)
            aB[mi] = load_a_frag_bf16(A + aoff[mi] + k0 + 32, hi);
#pragma unroll
        for (int ni = 0; ni < 4; ni++)
            bB[ni] = load_b_frag(Wt + boff[ni] + k0 + 32);
        __builtin_prefetch(A + aoff[0] + k0 + 512, 0, 3);
        __builtin_prefetch(A + aoff[1] + k0 + 512, 0, 3);
#pragma unroll
        for (int mi = 0; mi < 2; mi++)
#pragma unroll
            for (int ni = 0; ni < 4; ni++)
                acc[mi][ni] = wmma_bf16(aA[mi], bA[ni], acc[mi][ni]);

#pragma unroll
        for (int mi = 0; mi < 2; mi++)
            aA[mi] = load_a_frag_bf16(A + aoff[mi] + k0 + 64, hi);
#pragma unroll
        for (int ni = 0; ni < 4; ni++)
            bA[ni] = load_b_frag(Wt + boff[ni] + k0 + 64);
#pragma unroll
        for (int mi = 0; mi < 2; mi++)
#pragma unroll
            for (int ni = 0; ni < 4; ni++)
                acc[mi][ni] = wmma_bf16(aB[mi], bB[ni], acc[mi][ni]);
    }
    {
        v16bf aB[2], bB[4];
#pragma unroll
        for (int mi = 0; mi < 2; mi++)
            aB[mi] = load_a_frag_bf16(A + aoff[mi] + EE - 32, hi);
#pragma unroll
        for (int ni = 0; ni < 4; ni++)
            bB[ni] = load_b_frag(Wt + boff[ni] + EE - 32);
#pragma unroll
        for (int mi = 0; mi < 2; mi++)
#pragma unroll
            for (int ni = 0; ni < 4; ni++)
                acc[mi][ni] = wmma_bf16(aA[mi], bA[ni], acc[mi][ni]);
#pragma unroll
        for (int mi = 0; mi < 2; mi++)
#pragma unroll
            for (int ni = 0; ni < 4; ni++)
                acc[mi][ni] = wmma_bf16(aB[mi], bB[ni], acc[mi][ni]);
    }

#pragma unroll
    for (int mi = 0; mi < 2; mi++) {
#pragma unroll
        for (int j = 0; j < 8; j++) {
            const int m = m0 + mi * 16 + j + hi * 8;
            const float qm = (float)q_mask[m];
#pragma unroll
            for (int ni = 0; ni < 4; ni++) {
                const int n = n0 + ni * 16 + l;
                Out[(unsigned)(m * EE + n)] = (acc[mi][ni][j] + bias[n]) * qm;
            }
        }
    }
}

// ---------------------------------------------------------------------------
// Host-side launch
// ---------------------------------------------------------------------------
extern "C" void kernel_launch(void* const* d_in, const int* in_sizes, int n_in,
                              void* d_out, int out_size, void* d_ws, size_t ws_size,
                              hipStream_t stream) {
    const float* q      = (const float*)d_in[0];
    const float* k      = (const float*)d_in[1];
    const float* v      = (const float*)d_in[2];
    const int*   v_mask = (const int*)d_in[3];
    const int*   q_mask = (const int*)d_in[4];
    // d_in[5] = a_mask (causal tril) -- handled analytically in the kernel
    const float* Wq = (const float*)d_in[6];
    const float* bq = (const float*)d_in[7];
    const float* Wk = (const float*)d_in[8];
    const float* bk = (const float*)d_in[9];
    const float* Wv = (const float*)d_in[10];
    const float* bv = (const float*)d_in[11];
    const float* Wo = (const float*)d_in[12];
    const float* bo = (const float*)d_in[13];
    float* out = (float*)d_out;

    // Workspace layout (bf16 elements); total 48 MB
    const size_t W_ELEMS = (size_t)DDIM * EE;      // 1M per weight matrix
    const size_t T_ELEMS = (size_t)MM * EE;        // 4M per activation tensor
    unsigned short* ws  = (unsigned short*)d_ws;
    unsigned short* WtQ = ws;
    unsigned short* WtK = WtQ + W_ELEMS;
    unsigned short* WtV = WtK + W_ELEMS;
    unsigned short* WtO = WtV + W_ELEMS;
    unsigned short* Qp  = WtO + W_ELEMS;
    unsigned short* Kp  = Qp + T_ELEMS;
    unsigned short* Vt  = Kp + T_ELEMS;            // [B][H][DH][S]
    unsigned short* Oa  = Vt + T_ELEMS;
    unsigned short* Xb  = Oa + T_ELEMS;            // reused for q, then k, then v

    dim3 wg(DDIM / 32, EE / 32);                   // 32 x 32 tiles
    wtrans_kernel<<<wg, 256, 0, stream>>>(Wq, WtQ);
    wtrans_kernel<<<wg, 256, 0, stream>>>(Wk, WtK);
    wtrans_kernel<<<wg, 256, 0, stream>>>(Wv, WtV);
    wtrans_kernel<<<wg, 256, 0, stream>>>(Wo, WtO);

    const int cblocks = (int)(T_ELEMS / (256 * 8));
    dim3 g(MM / 128, EE / 128);                    // 32 x 8

    // Stream ordering makes the Xb reuse safe (kernels serialize on `stream`).
    cvt_bf16_kernel<<<cblocks, 256, 0, stream>>>(q, Xb);
    proj_gemm_kernel<<<g, 256, 0, stream>>>(Xb, WtQ, bq, Qp, 0);
    cvt_bf16_kernel<<<cblocks, 256, 0, stream>>>(k, Xb);
    proj_gemm_kernel<<<g, 256, 0, stream>>>(Xb, WtK, bk, Kp, 0);
    cvt_bf16_kernel<<<cblocks, 256, 0, stream>>>(v, Xb);
    proj_gemm_kernel<<<g, 256, 0, stream>>>(Xb, WtV, bv, Vt, 1);

    attn_kernel<<<(BB * HH * (SS / 16)) / 8, 256, 0, stream>>>(Qp, Kp, Vt, v_mask, Oa);

    out_gemm_kernel<<<g, 256, 0, stream>>>(Oa, WtO, bo, q_mask, out);
}
